// KeyFormer_18425409700520
// MI455X (gfx1250) — matrix operand order repaired and checked by reading, not compiled, not run on hardware
//
#include <hip/hip_runtime.h>
#include <hip/hip_bf16.h>

#define NH   32
#define QL   1024
#define KLEN 1024
#define DH   128
#define BQ   64
#define BK   32
#define SCALE 0.08838834764831845f   // 1/sqrt(128)
#define INV_TAU (1.0f/1.5f)
#define NEGINF (-__builtin_inff())

typedef _Float16 v16h __attribute__((ext_vector_type(16)));
typedef _Float16 v8h  __attribute__((ext_vector_type(8)));
typedef float    v8f  __attribute__((ext_vector_type(8)));
typedef unsigned int v4u __attribute__((ext_vector_type(4)));
typedef int v4i __attribute__((ext_vector_type(4)));
typedef int v8i __attribute__((ext_vector_type(8)));

union HV16 { v16h v; v8h h[2]; };

// cross-half (xor 16) exchange via ds_swizzle_b32
__device__ __forceinline__ float swz16(float x) {
  return __int_as_float(__builtin_amdgcn_ds_swizzle(__float_as_int(x), 0x401F));
}
// exp(s-m) with -inf guard (masked entries contribute 0, no -inf - -inf NaN)
__device__ __forceinline__ float pexp(float s, float m) {
  return (s == NEGINF) ? 0.0f : __expf(s - m);
}

// ---- TDM: async 2D tile load (32 cols x 64 rows, f32) global -> LDS
__device__ __forceinline__ void tdm_load_er_tile(const float* gp, unsigned lds_off) {
  unsigned long long ga = (unsigned long long)(const void*)gp;
  v4u g0;
  g0[0] = 1u;                                   // count=1 (valid user descriptor)
  g0[1] = lds_off;                              // lds_addr
  g0[2] = (unsigned)(ga & 0xFFFFFFFFu);         // global_addr[31:0]
  g0[3] = (unsigned)((ga >> 32) & 0x1FFFFFFu)   // global_addr[56:32]
        | 0x80000000u;                          // type=2 ("image")
  v8i g1;
  g1[0] = (int)(2u << 16);        // workgroup_mask=0, data_size=2 (4 bytes)
  g1[1] = (int)(1024u << 16);     // tensor_dim0 = KLEN (low 16 in [31:16])
  g1[2] = (int)(1024u << 16);     // tensor_dim0 hi=0 | tensor_dim1 = QL (low 16)
  g1[3] = (int)(32u << 16);       // tensor_dim1 hi=0 | tile_dim0 = 32
  g1[4] = 64;                     // tile_dim1 = 64, tile_dim2 = 0
  g1[5] = 1024;                   // tensor_dim0_stride = KLEN (low 32)
  g1[6] = 0;
  g1[7] = 0;
  v4i gz = {0, 0, 0, 0};
#if __clang_major__ >= 23
  v8i gz8 = {0, 0, 0, 0, 0, 0, 0, 0};
  __builtin_amdgcn_tensor_load_to_lds(g0, g1, gz, gz, gz8, 0);
#else
  __builtin_amdgcn_tensor_load_to_lds(g0, g1, gz, gz, 0);
#endif
}

__global__ void kf_zero_kernel(float* kf) {
  int i = blockIdx.x * blockDim.x + threadIdx.x;
  if (i < NH * KLEN) kf[i] = 0.0f;
}

__global__ __launch_bounds__(128)
void kf_attn_kernel(const float* __restrict__ q, const float* __restrict__ k,
                    const float* __restrict__ v, const float* __restrict__ er,
                    float* __restrict__ out, float* __restrict__ kf)
{
  __shared__ _Float16 Kt[BK * 136];               // [key][d]  stride 136 halfs
  __shared__ _Float16 Vt[DH * 40];                // [d][key]  stride 40 halfs
  __shared__ _Float16 Pp[4][16 * 40];             // per wave: [row][col] stride 40
  __shared__ __align__(16) float ErT[2][BQ * BK]; // TDM double buffer: [row][col]
  __shared__ float RedM[4][16];                   // per-wave row-max scratch
  __shared__ float RedS[4][16];                   // per-wave row-sum (attention)
  __shared__ float RedZ[4][16];                   // per-wave row-sum (gumbel)

  const int h     = blockIdx.x >> 4;
  const int qb    = blockIdx.x & 15;
  const int qbase = qb * BQ;
  const int tid   = threadIdx.x;
  const int wq    = tid >> 5;
  const int lane  = tid & 31;
  const int ln    = lane & 15;
  const int hi    = lane >> 4;
  const int ntiles = (qbase + BQ) / BK;           // causal tiles for this q-block

  const float* erbase = er + ((size_t)h * QL + qbase) * KLEN;
  const unsigned erlds0 = (unsigned)(unsigned long long)(const void*)&ErT[0][0];
  const unsigned erlds1 = (unsigned)(unsigned long long)(const void*)&ErT[1][0];

  // ---- Q fragments (A-matrix 16x32 f16 layout), kept across both passes
  const int qrowA = qbase + wq * 16 + ln;
  const float* qg = q + ((size_t)qrowA * NH + h) * DH;
  v16h aq[4];
  #pragma unroll
  for (int ks = 0; ks < 4; ++ks) {
    const int d0 = ks * 32;
    const int b  = hi * 16;
    HV16 t; v8h lo, hiv;
    #pragma unroll
    for (int i = 0; i < 8; ++i) lo[i]  = (_Float16)qg[d0 + b + i];
    #pragma unroll
    for (int i = 0; i < 8; ++i) hiv[i] = (_Float16)qg[d0 + b + 8 + i];
    t.h[0] = lo; t.h[1] = hiv;
    aq[ks] = t.v;
  }

  float m1[8], l1[8], l2[8];
  #pragma unroll
  for (int r = 0; r < 8; ++r) { m1[r] = NEGINF; l1[r] = 0.f; l2[r] = 0.f; }

  // =======================  PASS A: row statistics  =======================
  if (wq == 0) tdm_load_er_tile(erbase, erlds0);   // kick tile 0
  for (int kt = 0; kt < ntiles; ++kt) {
    __syncthreads();   // prior-iteration consumers finished
    for (int idx = tid * 4; idx < BK * DH; idx += 128 * 4) {
      const int kk = idx >> 7, dd = idx & 127;
      const float4 f = *(const float4*)&k[((size_t)(kt * BK + kk) * NH + h) * DH + dd];
      Kt[kk * 136 + dd + 0] = (_Float16)f.x;
      Kt[kk * 136 + dd + 1] = (_Float16)f.y;
      Kt[kk * 136 + dd + 2] = (_Float16)f.z;
      Kt[kk * 136 + dd + 3] = (_Float16)f.w;
      if (kt + 1 < ntiles)
        __builtin_prefetch(&k[((size_t)((kt + 1) * BK + kk) * NH + h) * DH + dd], 0, 3);
    }
    if (wq == 0) {
      if (kt + 1 < ntiles) {
        tdm_load_er_tile(erbase + (kt + 1) * BK, ((kt + 1) & 1) ? erlds1 : erlds0);
        __builtin_amdgcn_s_wait_tensorcnt(1);   // tile kt complete, kt+1 in flight
      } else {
        __builtin_amdgcn_s_wait_tensorcnt(0);
      }
    }
    __syncthreads();   // K tile + er tile visible

    // S = Q K^T for two 16-wide key sub-tiles
    v8f c0 = {0,0,0,0,0,0,0,0}, c1 = {0,0,0,0,0,0,0,0};
    #pragma unroll
    for (int ks = 0; ks < 4; ++ks) {
      const int d0 = ks * 32, koff = hi * 16;
      HV16 b0, b1;
      b0.h[0] = *(const v8h*)&Kt[(ln)      * 136 + d0 + koff];
      b0.h[1] = *(const v8h*)&Kt[(ln)      * 136 + d0 + koff + 8];
      b1.h[0] = *(const v8h*)&Kt[(16 + ln) * 136 + d0 + koff];
      b1.h[1] = *(const v8h*)&Kt[(16 + ln) * 136 + d0 + koff + 8];
      c0 = __builtin_amdgcn_wmma_f32_16x16x32_f16(false, aq[ks], false, b0.v, (short)0, c0, false, false);
      c1 = __builtin_amdgcn_wmma_f32_16x16x32_f16(false, aq[ks], false, b1.v, (short)0, c1, false, false);
    }

    // per-wave LDS scratch init (in-order DS within the wave, no barrier needed)
    if (hi == 0) { RedM[wq][ln] = NEGINF; RedS[wq][ln] = 0.f; RedZ[wq][ln] = 0.f; }

    const float* ert = &ErT[kt & 1][0];
    const int kc0 = kt * BK + ln, kc1 = kt * BK + 16 + ln;
    float s0v[8], s1v[8];
    #pragma unroll
    for (int r = 0; r < 8; ++r) {
      const int mrow = qbase + wq * 16 + hi * 8 + r;
      s0v[r] = (kc0 <= mrow) ? c0[r] * SCALE : NEGINF;
      s1v[r] = (kc1 <= mrow) ? c1[r] * SCALE : NEGINF;
      // 16-lane row max via native LDS float atomic (ds_max_num_f32)
      __hip_atomic_fetch_max(&RedM[wq][hi * 8 + r], fmaxf(s0v[r], s1v[r]),
                             __ATOMIC_RELAXED, __HIP_MEMORY_SCOPE_WAVEFRONT);
    }
    float mn_[8], corr_[8], corr2_[8];
    #pragma unroll
    for (int r = 0; r < 8; ++r) {
      const int row  = hi * 8 + r;
      const int rrow = wq * 16 + row;
      const int mrow = qbase + rrow;
      const float mt = RedM[wq][row];              // broadcast load (same addr/half)
      const float mn = fmaxf(m1[r], mt);
      corr_[r]  = (m1[r] == NEGINF) ? 0.f : __expf(m1[r] - mn);
      corr2_[r] = (m1[r] == NEGINF) ? 0.f : __expf((m1[r] - mn) * INV_TAU);
      // attention sum-exp via ds_add_f32
      __hip_atomic_fetch_add(&RedS[wq][row], pexp(s0v[r], mn) + pexp(s1v[r], mn),
                             __ATOMIC_RELAXED, __HIP_MEMORY_SCOPE_WAVEFRONT);
      // gumbel sum-exp, stabilizer mn/tau (valid since log(er) >= 0 => z <= s/tau)
      const float e0 = ert[rrow * BK + ln];
      const float e1 = ert[rrow * BK + 16 + ln];
      const float z0 = (kc0 <= mrow) ? (s0v[r] - __logf(e0)) * INV_TAU : NEGINF;
      const float z1 = (kc1 <= mrow) ? (s1v[r] - __logf(e1)) * INV_TAU : NEGINF;
      const float zb = mn * INV_TAU;
      __hip_atomic_fetch_add(&RedZ[wq][row], pexp(z0, zb) + pexp(z1, zb),
                             __ATOMIC_RELAXED, __HIP_MEMORY_SCOPE_WAVEFRONT);
      mn_[r] = mn;
    }
    #pragma unroll
    for (int r = 0; r < 8; ++r) {
      const int row = hi * 8 + r;
      l1[r] = l1[r] * corr_[r]  + RedS[wq][row];
      l2[r] = l2[r] * corr2_[r] + RedZ[wq][row];
      m1[r] = mn_[r];
    }
  }

  float rl1[8], rl2[8], m2[8];
  #pragma unroll
  for (int r = 0; r < 8; ++r) {
    rl1[r] = (l1[r] > 0.f) ? 1.f / l1[r] : 0.f;
    rl2[r] = (l2[r] > 0.f) ? 1.f / l2[r] : 0.f;
    m2[r]  = m1[r] * INV_TAU;                    // gumbel stabilizer
  }

  // =====================  PASS B: O = P V and kf sums  =====================
  v8f o[8];
  #pragma unroll
  for (int c = 0; c < 8; ++c) o[c] = (v8f){0,0,0,0,0,0,0,0};

  __syncthreads();                                  // pass-A readers done
  if (wq == 0) tdm_load_er_tile(erbase, erlds0);    // restart er pipeline
  for (int kt = 0; kt < ntiles; ++kt) {
    __syncthreads();
    for (int idx = tid * 4; idx < BK * DH; idx += 128 * 4) {
      const int kk = idx >> 7, dd = idx & 127;
      const float4 fk = *(const float4*)&k[((size_t)(kt * BK + kk) * NH + h) * DH + dd];
      Kt[kk * 136 + dd + 0] = (_Float16)fk.x;
      Kt[kk * 136 + dd + 1] = (_Float16)fk.y;
      Kt[kk * 136 + dd + 2] = (_Float16)fk.z;
      Kt[kk * 136 + dd + 3] = (_Float16)fk.w;
      const float4 fv = *(const float4*)&v[((size_t)(kt * BK + kk) * NH + h) * DH + dd];
      Vt[(dd + 0) * 40 + kk] = (_Float16)fv.x;      // transposed: [d][key]
      Vt[(dd + 1) * 40 + kk] = (_Float16)fv.y;
      Vt[(dd + 2) * 40 + kk] = (_Float16)fv.z;
      Vt[(dd + 3) * 40 + kk] = (_Float16)fv.w;
    }
    if (wq == 0) {
      if (kt + 1 < ntiles) {
        tdm_load_er_tile(erbase + (kt + 1) * BK, ((kt + 1) & 1) ? erlds1 : erlds0);
        __builtin_amdgcn_s_wait_tensorcnt(1);
      } else {
        __builtin_amdgcn_s_wait_tensorcnt(0);
      }
    }
    __syncthreads();

    v8f c0 = {0,0,0,0,0,0,0,0}, c1 = {0,0,0,0,0,0,0,0};
    #pragma unroll
    for (int ks = 0; ks < 4; ++ks) {
      const int d0 = ks * 32, koff = hi * 16;
      HV16 b0, b1;
      b0.h[0] = *(const v8h*)&Kt[(ln)      * 136 + d0 + koff];
      b0.h[1] = *(const v8h*)&Kt[(ln)      * 136 + d0 + koff + 8];
      b1.h[0] = *(const v8h*)&Kt[(16 + ln) * 136 + d0 + koff];
      b1.h[1] = *(const v8h*)&Kt[(16 + ln) * 136 + d0 + koff + 8];
      c0 = __builtin_amdgcn_wmma_f32_16x16x32_f16(false, aq[ks], false, b0.v, (short)0, c0, false, false);
      c1 = __builtin_amdgcn_wmma_f32_16x16x32_f16(false, aq[ks], false, b1.v, (short)0, c1, false, false);
    }

    _Float16* pp = &Pp[wq][0];
    const float* ert = &ErT[kt & 1][0];
    const int kc0 = kt * BK + ln, kc1 = kt * BK + 16 + ln;
    float col0 = 0.f, col1 = 0.f;
    #pragma unroll
    for (int r = 0; r < 8; ++r) {
      const int rrow = wq * 16 + hi * 8 + r;
      const int mrow = qbase + rrow;
      const float s0 = (kc0 <= mrow) ? c0[r] * SCALE : NEGINF;
      const float s1 = (kc1 <= mrow) ? c1[r] * SCALE : NEGINF;
      pp[(hi * 8 + r) * 40 + ln]      = (_Float16)(pexp(s0, m1[r]) * rl1[r]);
      pp[(hi * 8 + r) * 40 + 16 + ln] = (_Float16)(pexp(s1, m1[r]) * rl1[r]);
      const float e0 = ert[rrow * BK + ln];
      const float e1 = ert[rrow * BK + 16 + ln];
      const float z0 = (kc0 <= mrow) ? (s0 - __logf(e0)) * INV_TAU : NEGINF;
      const float z1 = (kc1 <= mrow) ? (s1 - __logf(e1)) * INV_TAU : NEGINF;
      col0 += pexp(z0, m2[r]) * rl2[r];
      col1 += pexp(z1, m2[r]) * rl2[r];
    }
    col0 += swz16(col0);    // combine rows 0-7 with 8-15 (xor 16)
    col1 += swz16(col1);
    __syncthreads();        // make Pp visible for A-layout reads
    if (hi == 0) {
      atomicAdd(&kf[h * KLEN + kt * BK + ln],      col0);
      atomicAdd(&kf[h * KLEN + kt * BK + 16 + ln], col1);
    }

    // O += P (16x32) x V (32x16 per d-chunk) — A-layout read of Pp
    HV16 pa;
    pa.h[0] = *(const v8h*)&pp[ln * 40 + hi * 16];
    pa.h[1] = *(const v8h*)&pp[ln * 40 + hi * 16 + 8];
    #pragma unroll
    for (int c = 0; c < 8; ++c) {
      HV16 bv;
      bv.h[0] = *(const v8h*)&Vt[(c * 16 + ln) * 40 + hi * 16];
      bv.h[1] = *(const v8h*)&Vt[(c * 16 + ln) * 40 + hi * 16 + 8];
      o[c] = __builtin_amdgcn_wmma_f32_16x16x32_f16(false, pa.v, false, bv.v, (short)0, o[c], false, false);
    }
  }

  // ---- write O
  #pragma unroll
  for (int c = 0; c < 8; ++c) {
    #pragma unroll
    for (int r = 0; r < 8; ++r) {
      const int mrow = qbase + wq * 16 + hi * 8 + r;
      out[((size_t)mrow * NH + h) * DH + c * 16 + ln] = o[c][r];
    }
  }
}

extern "C" void kernel_launch(void* const* d_in, const int* in_sizes, int n_in,
                              void* d_out, int out_size, void* d_ws, size_t ws_size,
                              hipStream_t stream) {
  (void)in_sizes; (void)n_in; (void)d_ws; (void)ws_size; (void)out_size;
  const float* q  = (const float*)d_in[0];
  const float* k  = (const float*)d_in[1];
  const float* v  = (const float*)d_in[2];
  const float* er = (const float*)d_in[3];
  float* out = (float*)d_out;
  float* kf  = out + (size_t)QL * NH * DH;

  kf_zero_kernel<<<(NH * KLEN + 255) / 256, 256, 0, stream>>>(kf);
  kf_attn_kernel<<<NH * (QL / BQ), 128, 0, stream>>>(q, k, v, er, out, kf);
}
